// GCN_75033078661563
// MI455X (gfx1250) — compile-verified, matrix operand order checked by reading, not run on hardware
//
#include <hip/hip_runtime.h>

#define N_NODES 50000
#define N_EDGES 800000
#define CH 64

typedef __attribute__((ext_vector_type(16))) __bf16 v16bf;
typedef __attribute__((ext_vector_type(8)))  __bf16 v8bf;
typedef __attribute__((ext_vector_type(8)))  float  v8f;

__device__ __forceinline__ void cvt_split(float v, __bf16& h, __bf16& l) {
    h = (__bf16)v;
    l = (__bf16)(v - (float)h);
}

// ---------------- degree / normalization ----------------

__global__ void k_deg_init(float* __restrict__ deg) {
    int i = blockIdx.x * blockDim.x + threadIdx.x;
    if (i < N_NODES) deg[i] = 1.0f;  // self-loop
}

__global__ void k_deg_count(const int* __restrict__ dst, float* __restrict__ deg) {
    int e = blockIdx.x * blockDim.x + threadIdx.x;
    if (e < N_EDGES) atomicAdd(&deg[dst[e]], 1.0f);
}

__global__ void k_dinv(const float* __restrict__ deg, float* __restrict__ dinv) {
    int i = blockIdx.x * blockDim.x + threadIdx.x;
    if (i < N_NODES) dinv[i] = rsqrtf(deg[i]);   // deg >= 1 always
}

// ---------------- GEMM: T = X @ W via bf16x3 WMMA (fp32-accurate) ----------------
// One wave per 16-row M tile; K = 64 (2 chunks of 32); N = 64 (4 tiles of 16).
// W is converted once per block into LDS, transposed (Wt[n][k]) in bf16 hi/lo
// planes, so each lane's B fragment is two aligned 16B ds_load_b128 runs.
// A layout (16-bit 16x32): lane m = lane%16, kbase = lane<16 ? 0 : 8,
//   element e -> k = chunk*32 + kbase + e + (e>=8 ? 8 : 0). B mirrors with n = lane%16.
// C/D f32 16x16: VGPR r -> row = r + (lane<16 ? 0 : 8), col = lane%16.

__global__ void __launch_bounds__(256)
k_gemm_bf16x3(const float* __restrict__ X,
              const float* __restrict__ W,
              float* __restrict__ T,
              int n_tiles) {
    __shared__ __align__(16) __bf16 sWhi[CH * CH];   // Wt_hi[n*64 + k]
    __shared__ __align__(16) __bf16 sWlo[CH * CH];   // Wt_lo[n*64 + k]

    // cooperative W convert+transpose (all waves participate, then barrier)
    for (int j = 0; j < (CH * CH) / 256; ++j) {
        int idx = threadIdx.x + j * 256;       // idx = k*64 + n (row-major W)
        int k = idx >> 6, n = idx & 63;
        __bf16 h, l;
        cvt_split(W[idx], h, l);
        sWhi[n * CH + k] = h;
        sWlo[n * CH + k] = l;
    }
    __syncthreads();

    const int lane = threadIdx.x & 31;
    const int wave = blockIdx.x * (blockDim.x >> 5) + (threadIdx.x >> 5);
    if (wave >= n_tiles) return;  // wave-uniform: EXEC all-ones inside

    const int m     = lane & 15;
    const int kbase = (lane < 16) ? 0 : 8;
    const float4* xr4 = (const float4*)(X + (size_t)(wave * 16 + m) * CH);

    // A fragments: two aligned 32B runs per chunk (k runs [kb, kb+8) and [kb+16, kb+24))
    v16bf ahi[2], alo[2];
#pragma unroll
    for (int c = 0; c < 2; ++c) {
        const int base = c * 32 + kbase;                 // multiple of 8
        float4 p[4];
        p[0] = xr4[(base >> 2) + 0];
        p[1] = xr4[(base >> 2) + 1];
        p[2] = xr4[((base + 16) >> 2) + 0];
        p[3] = xr4[((base + 16) >> 2) + 1];
        const float* pf = (const float*)p;
#pragma unroll
        for (int e = 0; e < 16; ++e) {
            __bf16 h, l;
            cvt_split(pf[e], h, l);
            ahi[c][e] = h;
            alo[c][e] = l;
        }
    }

#pragma unroll 1
    for (int nt = 0; nt < 4; ++nt) {
        const int col = nt * 16 + m;

        v16bf bhi[2], blo[2];
#pragma unroll
        for (int c = 0; c < 2; ++c) {
            const int base = col * CH + c * 32 + kbase;  // bf16 index, 16B aligned
            v8bf h0 = *(const v8bf*)(&sWhi[base]);
            v8bf h1 = *(const v8bf*)(&sWhi[base + 16]);
            v8bf l0 = *(const v8bf*)(&sWlo[base]);
            v8bf l1 = *(const v8bf*)(&sWlo[base + 16]);
#pragma unroll
            for (int e = 0; e < 8; ++e) {
                bhi[c][e]     = h0[e];
                bhi[c][e + 8] = h1[e];
                blo[c][e]     = l0[e];
                blo[c][e + 8] = l1[e];
            }
        }

        v8f acc = {};
#pragma unroll
        for (int c = 0; c < 2; ++c) {
            // low-order terms first, dominant term last (better summation)
            acc = __builtin_amdgcn_wmma_f32_16x16x32_bf16(
                      false, alo[c], false, bhi[c], (short)0, acc, false, false);
            acc = __builtin_amdgcn_wmma_f32_16x16x32_bf16(
                      false, ahi[c], false, blo[c], (short)0, acc, false, false);
            acc = __builtin_amdgcn_wmma_f32_16x16x32_bf16(
                      false, ahi[c], false, bhi[c], (short)0, acc, false, false);
        }

        const int rbase = (lane < 16) ? 0 : 8;
        float* out = T + (size_t)(wave * 16) * CH + col;
#pragma unroll
        for (int r = 0; r < 8; ++r)
            out[(size_t)(r + rbase) * CH] = acc[r];
    }
}

// ---------------- aggregation ----------------

// agg[v] = T[v] * dinv[v]^2   (self-loop contribution; fully initializes agg)
__global__ void k_self_init(const float4* __restrict__ T4, const float* __restrict__ dinv,
                            float4* __restrict__ agg4) {
    int i = blockIdx.x * blockDim.x + threadIdx.x;     // over N_NODES*16 float4s
    if (i < N_NODES * (CH / 4)) {
        float dv = dinv[i >> 4];
        float s = dv * dv;
        float4 t = T4[i];
        agg4[i] = make_float4(t.x * s, t.y * s, t.z * s, t.w * s);
    }
}

// agg[dst] += T[src] * dinv[src] * dinv[dst]
// 16 threads per edge, float4 gathers (global_load_b128), 4 f32 atomics each.
// Working set (12.8 MB) is resident in MI455X's 192 MB L2 -> atomics stay in L2.
__global__ void k_edge_scatter(const int* __restrict__ src, const int* __restrict__ dst,
                               const float* __restrict__ dinv, const float* __restrict__ T,
                               float* __restrict__ agg) {
    unsigned idx = blockIdx.x * blockDim.x + threadIdx.x;  // < N_EDGES*16 = 12.8M
    unsigned e  = idx >> 4;
    unsigned c4 = (idx & 15u) << 2;                        // channel base: 0,4,...,60
    if (e < N_EDGES) {
        int s = src[e], d = dst[e];
        float nrm = dinv[s] * dinv[d];
        float4 t = *(const float4*)(T + (size_t)s * CH + c4);
        float* a = agg + (size_t)d * CH + c4;
        atomicAdd(a + 0, t.x * nrm);
        atomicAdd(a + 1, t.y * nrm);
        atomicAdd(a + 2, t.z * nrm);
        atomicAdd(a + 3, t.w * nrm);
    }
}

__global__ void k_bias_relu(const float4* __restrict__ agg4, const float4* __restrict__ b4,
                            float4* __restrict__ h4) {
    int i = blockIdx.x * blockDim.x + threadIdx.x;
    if (i < N_NODES * (CH / 4)) {
        float4 v = agg4[i];
        float4 b = b4[i & 15];
        v.x = fmaxf(v.x + b.x, 0.0f);
        v.y = fmaxf(v.y + b.y, 0.0f);
        v.z = fmaxf(v.z + b.z, 0.0f);
        v.w = fmaxf(v.w + b.w, 0.0f);
        h4[i] = v;
    }
}

__global__ void k_bias(const float4* __restrict__ agg4, const float4* __restrict__ b4,
                       float4* __restrict__ out4) {
    int i = blockIdx.x * blockDim.x + threadIdx.x;
    if (i < N_NODES * (CH / 4)) {
        float4 v = agg4[i];
        float4 b = b4[i & 15];
        out4[i] = make_float4(v.x + b.x, v.y + b.y, v.z + b.z, v.w + b.w);
    }
}

// ---------------- launch ----------------

extern "C" void kernel_launch(void* const* d_in, const int* in_sizes, int n_in,
                              void* d_out, int out_size, void* d_ws, size_t ws_size,
                              hipStream_t stream) {
    const float* x   = (const float*)d_in[0];
    const int*   ei  = (const int*)d_in[1];
    const float* W1  = (const float*)d_in[2];
    const float* b1  = (const float*)d_in[3];
    const float* W2  = (const float*)d_in[4];
    const float* b2  = (const float*)d_in[5];
    float* out = (float*)d_out;

    const int* src = ei;              // edge_index[0]
    const int* dst = ei + N_EDGES;    // edge_index[1]

    // workspace layout (floats)
    float* ws   = (float*)d_ws;
    float* deg  = ws;                                 // 50000
    float* dinv = ws + 65536;                         // 50000
    float* bufA = ws + 131072;                        // 3.2M floats (16B aligned)
    float* bufB = bufA + (size_t)N_NODES * CH;        // 3.2M floats  (~26 MB total)

    const int THREADS = 256;
    const int g_nodes = (N_NODES + THREADS - 1) / THREADS;
    const int g_edges = (N_EDGES + THREADS - 1) / THREADS;
    const int g_vec4  = (N_NODES * (CH / 4) + THREADS - 1) / THREADS;       // 3125
    const unsigned g_scat = ((unsigned)N_EDGES * 16 + THREADS - 1) / THREADS; // 50000

    const int m_tiles = N_NODES / 16;                 // 3125 (exact)
    const int g_gemm  = (m_tiles + (THREADS / 32) - 1) / (THREADS / 32);

    // normalization
    k_deg_init <<<g_nodes, THREADS, 0, stream>>>(deg);
    k_deg_count<<<g_edges, THREADS, 0, stream>>>(dst, deg);
    k_dinv     <<<g_nodes, THREADS, 0, stream>>>(deg, dinv);

    // layer 1:  t1 = X@W1 (A) -> agg1 (B) -> h = relu(agg1+b1) (A)
    k_gemm_bf16x3 <<<g_gemm, THREADS, 0, stream>>>(x, W1, bufA, m_tiles);
    k_self_init   <<<g_vec4, THREADS, 0, stream>>>((const float4*)bufA, dinv, (float4*)bufB);
    k_edge_scatter<<<g_scat, THREADS, 0, stream>>>(src, dst, dinv, bufA, bufB);
    k_bias_relu   <<<g_vec4, THREADS, 0, stream>>>((const float4*)bufB, (const float4*)b1, (float4*)bufA);

    // layer 2:  t2 = h@W2 (B) -> agg2 (A) -> out = agg2 + b2
    k_gemm_bf16x3 <<<g_gemm, THREADS, 0, stream>>>(bufA, W2, bufB, m_tiles);
    k_self_init   <<<g_vec4, THREADS, 0, stream>>>((const float4*)bufB, dinv, (float4*)bufA);
    k_edge_scatter<<<g_scat, THREADS, 0, stream>>>(src, dst, dinv, bufB, bufA);
    k_bias        <<<g_vec4, THREADS, 0, stream>>>((const float4*)bufA, (const float4*)b2, (float4*)out);
}
